// DeformableCrossAttention_45483703664937
// MI455X (gfx1250) — matrix operand order repaired and checked by reading, not compiled, not run on hardware
//
#include <hip/hip_runtime.h>

// ---------------------------------------------------------------------------
// Deformable cross-attention for MI455X (gfx1250, wave32, WMMA).
// Pipeline:
//   1) prep: f32->bf16 weights; pack [W_off|W_attn|0] into one 256x128 weight.
//   2) WMMA GEMM: value = memory @ Wv + bv      (bf16 out, 64 MB -> fits L2)
//   3) WMMA GEMM: qproj = query @ Wq + bq       (offsets + attn logits)
//   4) sampling kernel: softmax + bilinear gather from bf16 value (L2-served)
//   5) WMMA GEMM: out = S @ Wout + bout         (f32 out)
// ---------------------------------------------------------------------------

typedef __bf16 bf16_t;
typedef __attribute__((ext_vector_type(8)))  float  v8f;
typedef __attribute__((ext_vector_type(8)))  bf16_t v8bf;
typedef __attribute__((ext_vector_type(16))) bf16_t v16bf;

#define NB     8
#define LQ     1024
#define CDIM   256
#define HH     128
#define WW     128
#define HWPIX  (HH * WW)

__device__ __forceinline__ unsigned short f2bfu(float f) {
  unsigned u = __builtin_bit_cast(unsigned, f);
  u += 0x7fffu + ((u >> 16) & 1u);            // round-to-nearest-even
  return (unsigned short)(u >> 16);
}
__device__ __forceinline__ bf16_t f2bf(float f) {
  return __builtin_bit_cast(bf16_t, f2bfu(f));
}
__device__ __forceinline__ float bf2f(bf16_t h) {
  unsigned u = ((unsigned)__builtin_bit_cast(unsigned short, h)) << 16;
  return __builtin_bit_cast(float, u);
}

// ---------------------------------------------------------------------------
// Tiled bf16 WMMA GEMM:  Cout[M,N] = A[M,K] @ B[K,N] + bias[N]
//   Block: 256 threads = 8 waves; tile BM=128, BN=128, BK=32, double-buffered.
//   Wave grid 4(M) x 2(N); each wave: 32x64 = 2x4 v_wmma_f32_16x16x32_bf16.
//   A staged to LDS as bf16 [row][k]; B staged TRANSPOSED as [col][k] so
//   every fragment is two contiguous 16B ds_load_b128 per the ISA layouts.
//   Double buffering: next tile's global loads are issued before this tile's
//   WMMA batch -> HBM latency hidden behind matrix ops; 1 barrier / K-step.
//   M, N must be multiples of 128; K a multiple of 32 (true for all calls).
// ---------------------------------------------------------------------------
template <typename AT, typename OT>
__global__ __launch_bounds__(256) void gemm_wmma_bf16(
    const AT* __restrict__ A, const bf16_t* __restrict__ B,
    const float* __restrict__ bias, OT* __restrict__ Cout,
    int M, int N, int K)
{
  constexpr int BK  = 32;
  constexpr int PAD = 40;   // 80B rows: 16B aligned, bank-spread (gcd(20,64)=4)
  __shared__ __align__(16) bf16_t As[2][128][PAD];   // 2 x 10 KB
  __shared__ __align__(16) bf16_t Bs[2][128][PAD];   // 2 x 10 KB (transposed)

  const int t      = threadIdx.x;
  const int lane   = t & 31;
  const int wave   = t >> 5;
  const int wmBase = (wave >> 1) << 5;   // 0,32,64,96
  const int wnBase = (wave & 1) << 6;    // 0,64
  const int m0     = blockIdx.y << 7;
  const int n0     = blockIdx.x << 7;
  const int lhalf  = lane >> 4;          // 0/1
  const int l15    = lane & 15;

  // ---- stage one (A,B) K-tile into LDS buffer `buf` ----
  auto stage = [&](int k0, int buf) {
    #pragma unroll
    for (int i = 0; i < 8; ++i) {
      int p   = t + (i << 8);
      int row = p >> 4;
      int kp  = (p & 15) << 1;
      if constexpr (sizeof(AT) == 4) {
        const float2 v = *(const float2*)(A + (size_t)(m0 + row) * K + k0 + kp);
        unsigned pk = ((unsigned)f2bfu(v.y) << 16) | (unsigned)f2bfu(v.x);
        *(unsigned*)&As[buf][row][kp] = pk;
      } else {
        unsigned pk = *(const unsigned*)((const bf16_t*)A +
                                         (size_t)(m0 + row) * K + k0 + kp);
        *(unsigned*)&As[buf][row][kp] = pk;
      }
    }
    #pragma unroll
    for (int i = 0; i < 8; ++i) {
      int p  = t + (i << 8);
      int kk = p >> 6;           // 0..31
      int cp = (p & 63) << 1;    // 0..126
      unsigned pk = *(const unsigned*)(B + (size_t)(k0 + kk) * N + n0 + cp);
      Bs[buf][cp][kk]     = __builtin_bit_cast(bf16_t, (unsigned short)(pk & 0xffffu));
      Bs[buf][cp + 1][kk] = __builtin_bit_cast(bf16_t, (unsigned short)(pk >> 16));
    }
  };

  v8f acc[2][4];
  #pragma unroll
  for (int mi = 0; mi < 2; ++mi)
    #pragma unroll
    for (int ni = 0; ni < 4; ++ni)
      #pragma unroll
      for (int r = 0; r < 8; ++r) acc[mi][ni][r] = 0.0f;

  stage(0, 0);
  __syncthreads();

  const int nk = K / BK;
  for (int kt = 0; kt < nk; ++kt) {
    const int buf = kt & 1;
    if (kt + 1 < nk) stage((kt + 1) * BK, buf ^ 1);   // prefetch next tile

    // ---- build fragments per ISA VGPR layouts ----
    // A 16x32: lanes 0-15 hold K {0..7,16..23}; lanes 16-31 hold {8..15,24..31}
    v16bf afrag[2], bfrag[4];
    const int kA = lhalf << 3;
    #pragma unroll
    for (int mi = 0; mi < 2; ++mi) {
      int row = wmBase + (mi << 4) + l15;
      v8bf lo = *(const v8bf*)&As[buf][row][kA];
      v8bf hi = *(const v8bf*)&As[buf][row][16 + kA];
      afrag[mi] = __builtin_shufflevector(lo, hi,
          0,1,2,3,4,5,6,7,8,9,10,11,12,13,14,15);
    }
    // B 32x16: lanes 0-15 -> K {0..7,16..23}; lanes 16-31 -> K {8..15,24..31} (col = lane&15)
    const int kB = lhalf << 3;
    #pragma unroll
    for (int ni = 0; ni < 4; ++ni) {
      int col = wnBase + (ni << 4) + l15;
      v8bf lo = *(const v8bf*)&Bs[buf][col][kB];
      v8bf hi = *(const v8bf*)&Bs[buf][col][16 + kB];
      bfrag[ni] = __builtin_shufflevector(lo, hi,
          0,1,2,3,4,5,6,7,8,9,10,11,12,13,14,15);
    }

    #pragma unroll
    for (int mi = 0; mi < 2; ++mi)
      #pragma unroll
      for (int ni = 0; ni < 4; ++ni)
        acc[mi][ni] = __builtin_amdgcn_wmma_f32_16x16x32_bf16(
            false, afrag[mi], false, bfrag[ni],
            (short)0, acc[mi][ni], false, false);

    __syncthreads();   // next-tile stores visible; safe to overwrite buf in kt+2
  }

  // ---- store: C/D layout: lane<16 -> M=r, N=lane; lane>=16 -> M=r+8 ----
  #pragma unroll
  for (int ni = 0; ni < 4; ++ni) {
    int col = n0 + wnBase + (ni << 4) + l15;
    float bv = bias[col];
    #pragma unroll
    for (int mi = 0; mi < 2; ++mi) {
      int rowBase = m0 + wmBase + (mi << 4) + (lhalf << 3);
      #pragma unroll
      for (int r = 0; r < 8; ++r) {
        float v = acc[mi][ni][r] + bv;
        size_t idx = (size_t)(rowBase + r) * N + col;
        if constexpr (sizeof(OT) == 2) {
          Cout[idx] = __builtin_bit_cast(OT, f2bfu(v));
        } else {
          Cout[idx] = v;
        }
      }
    }
  }
}

// ---------------------------------------------------------------------------
// Weight prep: f32 -> bf16; pack [W_off (64) | W_attn (32) | zeros (32)]
// into Wq [256 x 128] + bias vector bq[128] (b_off | b_attn | 0).
// ---------------------------------------------------------------------------
__global__ void prep_weights_kernel(
    const float* __restrict__ Wv,    const float* __restrict__ Woff,
    const float* __restrict__ Wattn, const float* __restrict__ boff,
    const float* __restrict__ battn, const float* __restrict__ Wout,
    bf16_t* __restrict__ Wv_bf, bf16_t* __restrict__ Wq_bf,
    bf16_t* __restrict__ Wout_bf, float* __restrict__ bq)
{
  int i = blockIdx.x * 256 + threadIdx.x;   // 0 .. 65535
  Wv_bf[i]   = f2bf(Wv[i]);
  Wout_bf[i] = f2bf(Wout[i]);
  if (i < 256 * 128) {
    int k = i >> 7, c = i & 127;
    float v = 0.0f;
    if (c < 64)      v = Woff[k * 64 + c];
    else if (c < 96) v = Wattn[k * 32 + (c - 64)];
    Wq_bf[i] = f2bf(v);
  }
  if (i < 128) {
    float v = 0.0f;
    if (i < 64)      v = boff[i];
    else if (i < 96) v = battn[i - 64];
    bq[i] = v;
  }
}

// ---------------------------------------------------------------------------
// Sampling: one block per (n, lq) row; thread t = output channel (h = t>>5).
// Softmax over 4 points, bilinear zero-padded gather from bf16 value tensor
// (value layout [N,HW,C]: corner fetch = 64 contiguous bytes per head, and
// the whole 64 MB value tensor is L2-resident on the 192 MB MI455X L2).
// ---------------------------------------------------------------------------
__global__ __launch_bounds__(256) void deform_sample_kernel(
    const bf16_t* __restrict__ value,   // [NB, HW, C] bf16
    const float*  __restrict__ qproj,   // [NB*LQ, 128]  (64 off | 32 attn | pad)
    const float*  __restrict__ refpts,  // [NB*LQ, 2]
    bf16_t* __restrict__ S)             // [NB*LQ, C] bf16
{
  const int row = blockIdx.x;
  const int n   = row >> 10;            // LQ = 1024
  const int t   = threadIdx.x;          // channel = h*32 + d = t
  const int h   = t >> 5;

  const float rx = refpts[row * 2 + 0];
  const float ry = refpts[row * 2 + 1];
  const float* qp = qproj + (size_t)row * 128;

  float l[4];
  #pragma unroll
  for (int p = 0; p < 4; ++p) l[p] = qp[64 + h * 4 + p];
  float mx = fmaxf(fmaxf(l[0], l[1]), fmaxf(l[2], l[3]));
  float e[4], es = 0.0f;
  #pragma unroll
  for (int p = 0; p < 4; ++p) { e[p] = __expf(l[p] - mx); es += e[p]; }
  const float inv = 1.0f / es;

  const bf16_t* vbase = value + (size_t)n * HWPIX * CDIM + t;

  float acc = 0.0f;
  #pragma unroll
  for (int p = 0; p < 4; ++p) {
    float ox = qp[h * 8 + p * 2 + 0];
    float oy = qp[h * 8 + p * 2 + 1];
    // x = loc_x * W - 0.5 ; loc = ref + off/[W,H]
    float x = (rx + ox * (1.0f / (float)WW)) * (float)WW - 0.5f;
    float y = (ry + oy * (1.0f / (float)HH)) * (float)HH - 0.5f;
    float x0f = floorf(x), y0f = floorf(y);
    int   x0 = (int)x0f,   y0 = (int)y0f;
    float wx1 = x - x0f, wx0 = 1.0f - wx1;
    float wy1 = y - y0f, wy0 = 1.0f - wy1;
    float aw = e[p] * inv;

    float s = 0.0f;
    #pragma unroll
    for (int cy = 0; cy < 2; ++cy) {
      int   yi = y0 + cy;
      float wy = cy ? wy1 : wy0;
      if (yi < 0 || yi >= HH) continue;
      #pragma unroll
      for (int cx = 0; cx < 2; ++cx) {
        int   xi = x0 + cx;
        float wx = cx ? wx1 : wx0;
        if (xi < 0 || xi >= WW) continue;
        float v = bf2f(vbase[(size_t)(yi * WW + xi) * CDIM]);
        s += wx * wy * v;
      }
    }
    acc += aw * s;
  }
  S[(size_t)row * CDIM + t] = f2bf(acc);
}

// ---------------------------------------------------------------------------
extern "C" void kernel_launch(void* const* d_in, const int* in_sizes, int n_in,
                              void* d_out, int out_size, void* d_ws, size_t ws_size,
                              hipStream_t stream) {
  const float* query  = (const float*)d_in[0];
  const float* memory = (const float*)d_in[1];
  const float* refpts = (const float*)d_in[2];
  const float* Wv     = (const float*)d_in[3];
  const float* bv     = (const float*)d_in[4];
  const float* Woff   = (const float*)d_in[5];
  const float* boff   = (const float*)d_in[6];
  const float* Wattn  = (const float*)d_in[7];
  const float* battn  = (const float*)d_in[8];
  const float* Wout   = (const float*)d_in[9];
  const float* bout   = (const float*)d_in[10];
  float* out = (float*)d_out;

  char* ws = (char*)d_ws;
  size_t off = 0;
  bf16_t* value_bf = (bf16_t*)(ws + off); off += (size_t)NB * HWPIX * CDIM * 2; // 64 MB
  float*  qproj    = (float*)(ws + off);  off += (size_t)NB * LQ * 128 * 4;     // 4 MB
  bf16_t* S_bf     = (bf16_t*)(ws + off); off += (size_t)NB * LQ * CDIM * 2;    // 4 MB
  bf16_t* Wv_bf    = (bf16_t*)(ws + off); off += (size_t)CDIM * CDIM * 2;
  bf16_t* Wout_bf  = (bf16_t*)(ws + off); off += (size_t)CDIM * CDIM * 2;
  bf16_t* Wq_bf    = (bf16_t*)(ws + off); off += (size_t)CDIM * 128 * 2;
  float*  bq       = (float*)(ws + off);  off += 512;

  prep_weights_kernel<<<256, 256, 0, stream>>>(Wv, Woff, Wattn, boff, battn, Wout,
                                               Wv_bf, Wq_bf, Wout_bf, bq);

  { // value = memory @ Wv + bv  -> bf16 [131072, 256]
    dim3 grid(CDIM / 128, (NB * HWPIX) / 128);
    gemm_wmma_bf16<float, bf16_t><<<grid, 256, 0, stream>>>(
        memory, Wv_bf, bv, value_bf, NB * HWPIX, CDIM, CDIM);
  }
  { // qproj = query @ Wq + bq  -> f32 [8192, 128]
    dim3 grid(1, (NB * LQ) / 128);
    gemm_wmma_bf16<float, float><<<grid, 256, 0, stream>>>(
        query, Wq_bf, bq, qproj, NB * LQ, 128, CDIM);
  }
  deform_sample_kernel<<<NB * LQ, 256, 0, stream>>>(value_bf, qproj, refpts, S_bf);
  { // out = S @ Wout + bout -> f32 [8192, 256]
    dim3 grid(CDIM / 128, (NB * LQ) / 128);
    gemm_wmma_bf16<bf16_t, float><<<grid, 256, 0, stream>>>(
        S_bf, Wout_bf, bout, out, NB * LQ, CDIM, CDIM);
  }
  (void)in_sizes; (void)n_in; (void)out_size; (void)ws_size;
}
